// Llama3Attention_32109175505325
// MI455X (gfx1250) — matrix operand order repaired and checked by reading, not compiled, not run on hardware
//
#include <hip/hip_runtime.h>
#include <stdint.h>

#define B_   2
#define S_   1024
#define HID_ 2048
#define H_   32
#define KV_  8
#define D_   64
#define G_   (H_ / KV_)

typedef __attribute__((ext_vector_type(16))) __bf16   bf16x16;
typedef __attribute__((ext_vector_type(8)))  float    f32x8;
typedef __attribute__((ext_vector_type(4)))  uint32_t u32x4;
typedef __attribute__((ext_vector_type(8)))  uint32_t u32x8;

union Frag { bf16x16 v; uint32_t u[8]; };

__device__ __forceinline__ uint32_t f32_to_bf16_bits(float f) {
  union { float f; uint32_t u; } x; x.f = f;
  uint32_t r = x.u + 0x7FFFu + ((x.u >> 16) & 1u);   // round-to-nearest-even
  return r >> 16;
}
__device__ __forceinline__ unsigned short f32_to_bf16(float f) {
  return (unsigned short)f32_to_bf16_bits(f);
}
__device__ __forceinline__ uint32_t pack_bf16(float lo, float hi) {
  return f32_to_bf16_bits(lo) | (f32_to_bf16_bits(hi) << 16);
}
__device__ __forceinline__ uint32_t lds_off(const void* p) {
  return (uint32_t)(uintptr_t)p;   // flat LDS address: low 32 bits = LDS byte offset
}

// ---------------------------------------------------------------------------
// Tensor Data Mover: 2D tile load, bf16 (data_size=2B), global row-major with
// row stride `stride0` (elements) -> contiguous LDS tile [tile_d1][tile_d0].
// D# per CDNA5 ISA: group0 = count/lds/global/type,
// group1 = mask/dsize/dims/tiles/strides. 2-operand form = <=2D tensor.
// ---------------------------------------------------------------------------
__device__ __forceinline__ void tdm_load_2d(uint32_t ldsaddr, const void* gptr,
                                            uint32_t tile_d0, uint32_t tile_d1,
                                            uint32_t stride0)
{
  uint64_t ga = (uint64_t)(uintptr_t)gptr;
  u32x4 g0;
  g0[0] = 1u;                                                 // count=1 (valid user D#)
  g0[1] = ldsaddr;                                            // lds_addr
  g0[2] = (uint32_t)ga;                                       // global_addr[31:0]
  g0[3] = ((uint32_t)(ga >> 32) & 0x01FFFFFFu) | (2u << 30);  // addr[56:32] | type=2
  u32x8 g1;
  g1[0] = 0x00010000u;                                        // wg_mask=0, data_size=1 (2B)
  g1[1] = (tile_d0 & 0xFFFFu) << 16;                          // tensor_dim0 = tile_d0
  g1[2] = (tile_d0 >> 16) | ((tile_d1 & 0xFFFFu) << 16);      // tensor_dim1 = tile_d1
  g1[3] = (tile_d1 >> 16) | (tile_d0 << 16);                  // tile_dim0
  g1[4] = tile_d1;                                            // tile_dim1, tile_dim2=0
  g1[5] = stride0;                                            // tensor_dim0_stride[31:0]
  g1[6] = 0u;
  g1[7] = 0u;
  asm volatile("tensor_load_to_lds %0, %1" :: "s"(g0), "s"(g1) : "memory");
}

// ---------------------------------------------------------------------------
// One-time f32 -> bf16 conversion (elementwise, n even).
// ---------------------------------------------------------------------------
__global__ void cvt_bf16_kernel(const float* __restrict__ in,
                                unsigned short* __restrict__ out, int n)
{
  int i = (blockIdx.x * blockDim.x + threadIdx.x) * 2;
  if (i >= n) return;
  *(uint32_t*)&out[i] = pack_bf16(in[i], in[i + 1]);
}

// W [K][N] f32 -> Wt [N][K] bf16, LDS-tiled 32x32 transpose.
__global__ __launch_bounds__(256)
void transpose_cvt_kernel(const float* __restrict__ W,
                          unsigned short* __restrict__ Wt, int K, int N)
{
  __shared__ unsigned short tile[32][33];
  const int k0 = blockIdx.y * 32, n0 = blockIdx.x * 32;
  const int tx = threadIdx.x, ty = threadIdx.y;   // 32 x 8
  for (int r = 0; r < 4; ++r) {
    int k = ty + r * 8;
    tile[k][tx] = f32_to_bf16(W[(size_t)(k0 + k) * N + n0 + tx]);
  }
  __syncthreads();
  for (int r = 0; r < 4; ++r) {
    int nr = ty + r * 8;
    Wt[(size_t)(n0 + nr) * K + k0 + tx] = tile[tx][nr];
  }
}

// Fused interleaved RoPE + bf16 convert on [B*S, nheads, D].
__global__ void rope_cvt_kernel(const float* __restrict__ x,
                                unsigned short* __restrict__ o,
                                int nheads, int total)
{
  int idx = blockIdx.x * blockDim.x + threadIdx.x;
  if (idx >= total) return;
  int j    = idx & (D_ / 2 - 1);
  int rem  = idx / (D_ / 2);
  int h    = rem % nheads;
  int srow = rem / nheads;                 // b*S + s
  int s    = srow & (S_ - 1);
  float freq = __expf(-(float)j * (9.2103403720f / 32.0f));  // 10000^(-2j/D)
  float sn, cs; __sincosf((float)s * freq, &sn, &cs);
  const float* p = x + ((size_t)srow * nheads + h) * D_ + 2 * j;
  float xe = p[0], xo = p[1];
  *(uint32_t*)&o[((size_t)srow * nheads + h) * D_ + 2 * j] =
      pack_bf16(xe * cs - xo * sn, xo * cs + xe * sn);
}

// V f32 [b][s][kv][d] -> Vt bf16 [b][kv][d][s]  (TDM-friendly, key-contiguous)
__global__ void vt_cvt_kernel(const float* __restrict__ V,
                              unsigned short* __restrict__ Vt, int total)
{
  int idx = blockIdx.x * blockDim.x + threadIdx.x;
  if (idx >= total) return;
  int s   = idx & (S_ - 1);
  int rem = idx >> 10;
  int d   = rem & (D_ - 1); rem >>= 6;
  int kv  = rem & (KV_ - 1);
  int b   = rem >> 3;
  Vt[idx] = f32_to_bf16(V[(((size_t)b * S_ + s) * KV_ + kv) * D_ + d]);
}

// ---------------------------------------------------------------------------
// GEMM: Y[M,N] = A[M,K](bf16) @ Bt[N,K]^T(bf16), f32 out. 128 thr = 4 waves,
// 64x64 Y tile; wave 0 drives the TDM for both LDS tiles each K-step.
// ---------------------------------------------------------------------------
__global__ __launch_bounds__(128)
void gemm_bf16_tdm(const unsigned short* __restrict__ A,
                   const unsigned short* __restrict__ Bt,
                   float* __restrict__ Y, int M, int N, int K)
{
  __shared__ __attribute__((aligned(16))) unsigned short at[64][32];  // [m][k]
  __shared__ __attribute__((aligned(16))) unsigned short bt[64][32];  // [n][k]

  const int t    = threadIdx.x;
  const int lane = t & 31;
  const int half = lane >> 4;
  const int lr   = lane & 15;
  const int wave = t >> 5;
  const int m0   = blockIdx.y * 64;
  const int n0   = blockIdx.x * 64;
  const int msub = (wave & 1) * 32;
  const int nsub = (wave >> 1) * 32;

  f32x8 zero = {};
  f32x8 acc[2][2];
  for (int mi = 0; mi < 2; ++mi)
    for (int ni = 0; ni < 2; ++ni) acc[mi][ni] = zero;

  for (int kk = 0; kk < K; kk += 32) {
    if (t < 32) {   // wave 0 only: one DMA per tile per workgroup
      tdm_load_2d(lds_off(&at[0][0]), A  + (size_t)m0 * K + kk, 32, 64, (uint32_t)K);
      tdm_load_2d(lds_off(&bt[0][0]), Bt + (size_t)n0 * K + kk, 32, 64, (uint32_t)K);
      __builtin_amdgcn_s_wait_tensorcnt(0);
    }
    __syncthreads();

    Frag a[2], b[2];
    for (int mi = 0; mi < 2; ++mi)
      for (int i = 0; i < 8; ++i) {
        int k0 = ((i < 4) ? (2 * i) : (16 + 2 * (i - 4))) + 8 * half;
        a[mi].u[i] = *(const uint32_t*)&at[msub + mi * 16 + lr][k0];
      }
    for (int ni = 0; ni < 2; ++ni)
      for (int i = 0; i < 8; ++i) {
        int c0 = ((i < 4) ? (2 * i) : (16 + 2 * (i - 4))) + 8 * half;
        b[ni].u[i] = *(const uint32_t*)&bt[nsub + ni * 16 + lr][c0];
      }
    for (int mi = 0; mi < 2; ++mi)
      for (int ni = 0; ni < 2; ++ni)
        acc[mi][ni] = __builtin_amdgcn_wmma_f32_16x16x32_bf16(
            false, a[mi].v, false, b[ni].v, (short)0, acc[mi][ni], false, false);
    __syncthreads();
  }

  for (int mi = 0; mi < 2; ++mi)
    for (int ni = 0; ni < 2; ++ni)
      for (int j = 0; j < 8; ++j) {
        int row = m0 + msub + mi * 16 + j + 8 * half;
        int col = n0 + nsub + ni * 16 + lr;
        Y[(size_t)row * N + col] = acc[mi][ni][j];
      }
}

// ---------------------------------------------------------------------------
// Flash attention, causal, GQA. One wave per (b, h, 16-query tile).
// K/V tiles staged via TDM; scores and P*V via bf16 WMMA; f32 online softmax.
// ---------------------------------------------------------------------------
__global__ __launch_bounds__(32)
void flash_attn_wmma(const unsigned short* __restrict__ Qbf,  // [b][s][h][d]
                     const unsigned short* __restrict__ Kbf,  // [b][s][kv][d]
                     const unsigned short* __restrict__ Vt,   // [b][kv][d][s]
                     unsigned short* __restrict__ Cbf)        // [b][s][h][d]
{
  __shared__ __attribute__((aligned(16))) unsigned short kt[32][64];  // [key][d]
  __shared__ __attribute__((aligned(16))) unsigned short vs[64][32];  // [d][key]
  __shared__ __attribute__((aligned(16))) unsigned short pt[16][32];  // [q][key]

  const int lane = threadIdx.x & 31;
  const int half = lane >> 4;
  const int lr   = lane & 15;

  int id = blockIdx.x;
  const int qb = id & (S_ / 16 - 1); id >>= 6;
  const int h  = id & (H_ - 1);      id >>= 5;
  const int b  = id;
  const int kvh = h / G_;

  Frag qf[2];
  const unsigned short* qrow = Qbf + (((size_t)b * S_ + qb * 16 + lr) * H_ + h) * D_;
  for (int c = 0; c < 2; ++c)
    for (int i = 0; i < 8; ++i) {
      int k0 = ((i < 4) ? (2 * i) : (16 + 2 * (i - 4))) + 8 * half + 32 * c;
      qf[c].u[i] = *(const uint32_t*)&qrow[k0];
    }

  f32x8 zero = {};
  f32x8 o[4];
  for (int n = 0; n < 4; ++n) o[n] = zero;
  float mrow[8], lsum[8];
  for (int j = 0; j < 8; ++j) { mrow[j] = -1e30f; lsum[j] = 0.0f; }
  const float scale = 0.125f;            // 1/sqrt(64)
  const int qmax = qb * 16 + 15;

  for (int kb = 0; kb <= qmax; kb += 32) {
    // K tile: 32 keys x 64 d  (row stride KV*D);  V tile: 64 d x 32 keys (row stride S)
    tdm_load_2d(lds_off(&kt[0][0]),
                Kbf + (((size_t)b * S_ + kb) * KV_ + kvh) * D_, 64, 32, KV_ * D_);
    tdm_load_2d(lds_off(&vs[0][0]),
                Vt + ((size_t)(b * KV_ + kvh) * D_) * S_ + kb, 32, 64, S_);
    __builtin_amdgcn_s_wait_tensorcnt(0);
    __syncthreads();

    f32x8 s0 = zero, s1 = zero;
    for (int c = 0; c < 2; ++c) {
      Frag b0, b1;
      for (int i = 0; i < 8; ++i) {
        int d0 = ((i < 4) ? (2 * i) : (16 + 2 * (i - 4))) + 8 * half + 32 * c;
        b0.u[i] = *(const uint32_t*)&kt[lr][d0];
        b1.u[i] = *(const uint32_t*)&kt[16 + lr][d0];
      }
      s0 = __builtin_amdgcn_wmma_f32_16x16x32_bf16(false, qf[c].v, false, b0.v, (short)0, s0, false, false);
      s1 = __builtin_amdgcn_wmma_f32_16x16x32_bf16(false, qf[c].v, false, b1.v, (short)0, s1, false, false);
    }

    for (int j = 0; j < 8; ++j) {
      const int q = qb * 16 + j + 8 * half;
      float a0 = s0[j] * scale;
      float a1 = s1[j] * scale;
      if (kb + lr      > q) a0 = -1e30f;
      if (kb + 16 + lr > q) a1 = -1e30f;
      float mx = fmaxf(a0, a1);
      for (int off = 8; off >= 1; off >>= 1)
        mx = fmaxf(mx, __shfl_xor(mx, off, 16));
      float mn   = fmaxf(mrow[j], mx);
      float corr = __expf(mrow[j] - mn);
      mrow[j] = mn;
      float p0 = __expf(a0 - mn);
      float p1 = __expf(a1 - mn);
      float rs = p0 + p1;
      for (int off = 8; off >= 1; off >>= 1)
        rs += __shfl_xor(rs, off, 16);
      lsum[j] = lsum[j] * corr + rs;
      for (int n = 0; n < 4; ++n) o[n][j] *= corr;
      pt[j + 8 * half][lr]      = f32_to_bf16(p0);
      pt[j + 8 * half][16 + lr] = f32_to_bf16(p1);
    }
    __syncthreads();

    Frag pf;
    for (int i = 0; i < 8; ++i) {
      int k0 = ((i < 4) ? (2 * i) : (16 + 2 * (i - 4))) + 8 * half;
      pf.u[i] = *(const uint32_t*)&pt[lr][k0];
    }
    for (int n = 0; n < 4; ++n) {
      Frag vf;
      for (int i = 0; i < 8; ++i) {
        int c0 = ((i < 4) ? (2 * i) : (16 + 2 * (i - 4))) + 8 * half;
        vf.u[i] = *(const uint32_t*)&vs[n * 16 + lr][c0];
      }
      o[n] = __builtin_amdgcn_wmma_f32_16x16x32_bf16(false, pf.v, false, vf.v, (short)0, o[n], false, false);
    }
    __syncthreads();
  }

  for (int n = 0; n < 4; ++n)
    for (int j = 0; j < 8; ++j) {
      int q = qb * 16 + j + 8 * half;
      int d = n * 16 + lr;
      Cbf[(((size_t)b * S_ + q) * H_ + h) * D_ + d] = f32_to_bf16(o[n][j] / lsum[j]);
    }
}

// ---------------------------------------------------------------------------
extern "C" void kernel_launch(void* const* d_in, const int* in_sizes, int n_in,
                              void* d_out, int out_size, void* d_ws, size_t ws_size,
                              hipStream_t stream)
{
  (void)in_sizes; (void)n_in; (void)out_size; (void)ws_size;
  const float* X  = (const float*)d_in[0];
  const float* Wq = (const float*)d_in[1];
  const float* Wk = (const float*)d_in[2];
  const float* Wv = (const float*)d_in[3];
  const float* Wo = (const float*)d_in[4];
  float* out = (float*)d_out;

  const int M   = B_ * S_;                 // 2048
  const size_t MHD = (size_t)M * H_ * D_;  // 4.19M
  const size_t MKD = (size_t)M * KV_ * D_; // 1.05M

  float* Qf = (float*)d_ws;                // f32 intermediates
  float* Kf = Qf + MHD;
  float* Vf = Kf + MKD;
  unsigned short* Xbf = (unsigned short*)(Vf + MKD);   // bf16 buffers
  unsigned short* WqT = Xbf + (size_t)M * HID_;
  unsigned short* WkT = WqT + (size_t)HID_ * (H_ * D_);
  unsigned short* WvT = WkT + (size_t)HID_ * (KV_ * D_);
  unsigned short* WoT = WvT + (size_t)HID_ * (KV_ * D_);
  unsigned short* Qbf = WoT + (size_t)(H_ * D_) * HID_;
  unsigned short* Kbf = Qbf + MHD;
  unsigned short* Vtb = Kbf + MKD;
  unsigned short* Cbf = Vtb + MKD;

  // One-time bf16 conversions / transposes (bandwidth-bound pre-passes).
  const int nX = M * HID_;
  cvt_bf16_kernel<<<nX / 512, 256, 0, stream>>>(X, Xbf, nX);
  transpose_cvt_kernel<<<dim3((H_ * D_) / 32, HID_ / 32), dim3(32, 8), 0, stream>>>(Wq, WqT, HID_, H_ * D_);
  transpose_cvt_kernel<<<dim3((KV_ * D_) / 32, HID_ / 32), dim3(32, 8), 0, stream>>>(Wk, WkT, HID_, KV_ * D_);
  transpose_cvt_kernel<<<dim3((KV_ * D_) / 32, HID_ / 32), dim3(32, 8), 0, stream>>>(Wv, WvT, HID_, KV_ * D_);
  transpose_cvt_kernel<<<dim3(HID_ / 32, (H_ * D_) / 32), dim3(32, 8), 0, stream>>>(Wo, WoT, H_ * D_, HID_);

  // QKV projections (f32 accumulate).
  gemm_bf16_tdm<<<dim3((H_ * D_) / 64,  M / 64), 128, 0, stream>>>(Xbf, WqT, Qf, M, H_ * D_,  HID_);
  gemm_bf16_tdm<<<dim3((KV_ * D_) / 64, M / 64), 128, 0, stream>>>(Xbf, WkT, Kf, M, KV_ * D_, HID_);
  gemm_bf16_tdm<<<dim3((KV_ * D_) / 64, M / 64), 128, 0, stream>>>(Xbf, WvT, Vf, M, KV_ * D_, HID_);

  // RoPE + bf16 pack; V transpose to [b][kv][d][s].
  const int totq = M * H_  * (D_ / 2);
  const int totk = M * KV_ * (D_ / 2);
  const int totv = M * KV_ * D_;
  rope_cvt_kernel<<<(totq + 255) / 256, 256, 0, stream>>>(Qf, Qbf, H_,  totq);
  rope_cvt_kernel<<<(totk + 255) / 256, 256, 0, stream>>>(Kf, Kbf, KV_, totk);
  vt_cvt_kernel<<<(totv + 255) / 256, 256, 0, stream>>>(Vf, Vtb, totv);

  flash_attn_wmma<<<B_ * H_ * (S_ / 16), 32, 0, stream>>>(Qbf, Kbf, Vtb, Cbf);

  gemm_bf16_tdm<<<dim3(HID_ / 64, M / 64), 128, 0, stream>>>(Cbf, WoT, out, M, HID_, HID_);
}
